// CompLayer_79611513798971
// MI455X (gfx1250) — matrix operand order repaired and checked by reading, not compiled, not run on hardware
//
#include <hip/hip_runtime.h>
#include <hip/hip_bf16.h>

// ---------------------------------------------------------------------------
// Problem constants: B=4, T=16, C=1024, H=16, D=64, N2=512, OUT=32,
// S=4096, Nt=256.
// ---------------------------------------------------------------------------
#define BB   4
#define SS   4096
#define CC   1024
#define HH   16
#define DD   64
#define NN2  512
#define TT   16
#define NT   256
#define OUTD 32
#define C2   2048

typedef __attribute__((ext_vector_type(16))) __bf16 v16bf;
typedef __attribute__((ext_vector_type(8)))  float  v8f;
typedef __attribute__((ext_vector_type(4)))  unsigned int u32x4;
typedef __attribute__((ext_vector_type(8)))  int i32x8;
typedef __attribute__((ext_vector_type(4)))  int i32x4;

#if __has_builtin(__builtin_amdgcn_tensor_load_to_lds)
#define HAVE_TDM 1
#else
#define HAVE_TDM 0
#endif

// float -> bf16 bits, round-to-nearest-even
__device__ __forceinline__ unsigned short f2bf(float f) {
  unsigned int u = __builtin_bit_cast(unsigned int, f);
  unsigned int r = (u + 0x7FFFu + ((u >> 16) & 1u)) >> 16;
  return (unsigned short)r;
}
__device__ __forceinline__ __bf16 bfbits(unsigned short s) {
  return __builtin_bit_cast(__bf16, s);
}

// A-matrix (16x32 bf16) per-lane K map (ISA 7.12.2)
__device__ __forceinline__ int amapK(int i, int hi) {
  return (i < 8 ? i : i + 8) + hi * 8;
}

__device__ __forceinline__ v8f wmma_bf16(v16bf a, v16bf b, v8f c) {
  return __builtin_amdgcn_wmma_f32_16x16x32_bf16(false, a, false, b, (short)0,
                                                 c, false, false);
}

// low 32 bits of a generic pointer to __shared__ == LDS byte offset
__device__ __forceinline__ unsigned lds_off(const void* p) {
  return (unsigned)(unsigned long long)(uintptr_t)p;
}

// async 16-byte global -> LDS copy (ASYNCcnt-tracked)
__device__ __forceinline__ void async_b128(unsigned lds, unsigned long long ga) {
  asm volatile("global_load_async_to_lds_b128 %0, %1, off"
               :: "v"(lds), "v"(ga) : "memory");
}
__device__ __forceinline__ void wait_async0() {
  asm volatile("s_wait_asynccnt 0x0" ::: "memory");
}

#if HAVE_TDM
// TDM: DMA a 2-D tile (tile_d0 x tile_d1 bf16) of a row-major tensor with row
// length `stride` (elements) into LDS at `ldsaddr`, padding +16B every 128B so
// rows land with LDS stride = tile_d0*2 + 16 bytes.
__device__ __forceinline__ void tdm_load_2d_bf16(unsigned ldsaddr,
                                                 unsigned long long gaddr,
                                                 int tile_d0, int tile_d1,
                                                 int dim0, int dim1,
                                                 int stride) {
  u32x4 g0 = {1u,                 // count=1, user mode
              ldsaddr,            // lds_addr [63:32]
              (unsigned)gaddr,    // global_addr low
              (unsigned)(gaddr >> 32) | 0x80000000u};  // addr hi | type=2
  i32x8 g1 = {
      (int)((1u << 16)            // data_size = 2 bytes
            | (1u << 20)          // pad_enable
            | (4u << 22)          // pad_interval: 32 DWORDs (128B)
            | (3u << 25)),        // pad_amount: 4 DWORDs (16B)
      (int)(((unsigned)dim0 & 0xFFFFu) << 16),                 // tensor_dim0 lo
      (int)(((unsigned)dim0 >> 16) | (((unsigned)dim1 & 0xFFFFu) << 16)),
      (int)(((unsigned)dim1 >> 16) | ((unsigned)tile_d0 << 16)),  // tile_dim0
      (int)((unsigned)tile_d1),                                // tile_dim1, d2=0
      (int)stride,                                             // dim0_stride lo
      0, 0};
  i32x4 z4 = {0, 0, 0, 0};
#if defined(__clang_major__) && (__clang_major__ >= 23)
  i32x8 z8 = {0, 0, 0, 0, 0, 0, 0, 0};
  __builtin_amdgcn_tensor_load_to_lds(g0, g1, z4, z4, z8, 0);
#else
  __builtin_amdgcn_tensor_load_to_lds(g0, g1, z4, z4, 0);
#endif
}
#endif

// ---------------------------------------------------------------------------
// fp32 -> bf16 bulk convert (8 elems/thread)
// ---------------------------------------------------------------------------
__global__ __launch_bounds__(256) void cvt_kernel(const float* __restrict__ in,
                                                  unsigned short* __restrict__ out,
                                                  int n8) {
  int i = blockIdx.x * blockDim.x + threadIdx.x;
  if (i >= n8) return;
  float4 a = ((const float4*)in)[i * 2];
  float4 b = ((const float4*)in)[i * 2 + 1];
  uint4 u;
  u.x = (unsigned)f2bf(a.x) | ((unsigned)f2bf(a.y) << 16);
  u.y = (unsigned)f2bf(a.z) | ((unsigned)f2bf(a.w) << 16);
  u.z = (unsigned)f2bf(b.x) | ((unsigned)f2bf(b.y) << 16);
  u.w = (unsigned)f2bf(b.z) | ((unsigned)f2bf(b.w) << 16);
  ((uint4*)out)[i] = u;
}

// ---------------------------------------------------------------------------
// Tiled bf16 GEMM: Out[M,N] = A[M,K] @ Bw[K,N] + bias (+ Res)
// A,Bw are bf16 bits. Tile 128x64, BK=32, 256 thr = 8 waves (4x2),
// wave owns 32x32 = 2x2 frags. A tile staged via async-to-LDS; B (weight)
// tile staged via the Tensor Data Mover (fallback: async loads).
// Assumes M%128==0, N%64==0, K%32==0.
// ---------------------------------------------------------------------------
template <bool RES, bool OUTBF>
__global__ __launch_bounds__(256) void gemm_wmma_kernel(
    const unsigned short* __restrict__ A, const unsigned short* __restrict__ Bw,
    const float* __restrict__ bias, const float* __restrict__ Res,
    void* __restrict__ Out, int M, int N, int K) {
  __shared__ unsigned short As[128][40];  // 128x32 (+8 pad) : row 80 B
  __shared__ unsigned short Bs[32][72];   // 32x64 (+8 pad)  : row 144 B

  const int tid  = threadIdx.x;
  const int lane = tid & 31;
  const int wave = tid >> 5;
  const int wm   = wave >> 1;
  const int wn   = wave & 1;
  const int l16  = lane & 15;
  const int hi   = lane >> 4;
  const int bm   = blockIdx.y * 128;
  const int bn   = blockIdx.x * 64;

  const unsigned asb = lds_off(&As[0][0]);
  const unsigned bsb = lds_off(&Bs[0][0]);
  const unsigned long long abase = (unsigned long long)(uintptr_t)A;
  const unsigned long long bbase = (unsigned long long)(uintptr_t)Bw;

  v8f acc[2][2] = {};

  for (int k0 = 0; k0 < K; k0 += 32) {
    __syncthreads();
    // A tile: 128 rows x 64B = 512 x 16B transfers, 2 per thread (async)
#pragma unroll
    for (int j = 0; j < 2; ++j) {
      int tr = tid * 2 + j;
      int r = tr >> 2, seg = tr & 3;
      async_b128(asb + r * 80 + seg * 16,
                 abase + ((unsigned long long)(bm + r) * K + k0) * 2 + seg * 16);
    }
    // B tile: 32x64 bf16 tile of (K x N) weight tensor
#if HAVE_TDM
    if (wave == 0) {
      tdm_load_2d_bf16(bsb, bbase + ((unsigned long long)k0 * N + bn) * 2,
                       /*tile*/ 64, 32, /*tensor*/ N, K, /*stride*/ N);
    }
#else
    {
      int r = tid >> 3, seg = tid & 7;  // 256 x 16B transfers, 1 per thread
      async_b128(bsb + r * 144 + seg * 16,
                 bbase + ((unsigned long long)(k0 + r) * N + bn) * 2 + seg * 16);
    }
#endif
    wait_async0();
#if HAVE_TDM
    if (wave == 0) __builtin_amdgcn_s_wait_tensorcnt(0);
#endif
    __syncthreads();

    v16bf af[2], bfr[2];
#pragma unroll
    for (int fm = 0; fm < 2; ++fm) {
      const int r0 = wm * 32 + fm * 16 + l16;
#pragma unroll
      for (int i = 0; i < 16; ++i) af[fm][i] = bfbits(As[r0][amapK(i, hi)]);
    }
#pragma unroll
    for (int fn = 0; fn < 2; ++fn) {
      const int c0 = wn * 32 + fn * 16 + l16;
#pragma unroll
      for (int i = 0; i < 16; ++i) bfr[fn][i] = bfbits(Bs[amapK(i, hi)][c0]);
    }
#pragma unroll
    for (int fm = 0; fm < 2; ++fm)
#pragma unroll
      for (int fn = 0; fn < 2; ++fn)
        acc[fm][fn] = wmma_bf16(af[fm], bfr[fn], acc[fm][fn]);
  }

  // Epilogue. C/D layout: VGPR v -> M = v + 8*hi, N = lane%16.
#pragma unroll
  for (int fm = 0; fm < 2; ++fm) {
#pragma unroll
    for (int fn = 0; fn < 2; ++fn) {
      const int gn = bn + wn * 32 + fn * 16 + l16;
      const float bv = bias[gn];
#pragma unroll
      for (int v = 0; v < 8; ++v) {
        const int gm = bm + wm * 32 + fm * 16 + v + hi * 8;
        float val = acc[fm][fn][v] + bv;
        if (RES) val += Res[(size_t)gm * N + gn];
        if (OUTBF)
          ((unsigned short*)Out)[(size_t)gm * N + gn] = f2bf(val);
        else
          ((float*)Out)[(size_t)gm * N + gn] = val;
      }
    }
  }
}

// ---------------------------------------------------------------------------
// Cross-attention, one wave per (b, h, 16-row q tile). N2=512, D=64.
// Tile staging via async global->LDS; scores/softmax fp32 in LDS; WMMA bf16.
// ---------------------------------------------------------------------------
__global__ __launch_bounds__(32) void attn_kernel(
    const unsigned short* __restrict__ qb,
    const unsigned short* __restrict__ kvb,
    unsigned short* __restrict__ ao) {
  __shared__ float          sc[16][512];
  __shared__ unsigned short pl[16][512];
  __shared__ unsigned short kvst[32][72];  // row 144 B
  __shared__ unsigned short qst[16][72];

  const int lane = threadIdx.x;
  const int l16  = lane & 15;
  const int hi   = lane >> 4;
  const int qt = blockIdx.x;
  const int h  = blockIdx.y;
  const int b  = blockIdx.z;
  const int s0 = qt * 16;

  const unsigned qstb = lds_off(&qst[0][0]);
  const unsigned kvlb = lds_off(&kvst[0][0]);
  const unsigned long long qgb =
      (unsigned long long)(uintptr_t)qb +
      ((unsigned long long)((size_t)b * SS + s0) * CC + h * DD) * 2;
  const unsigned long long kvgb = (unsigned long long)(uintptr_t)kvb;

  // Stage q tile 16 x 128B : 128 transfers, 4/lane
#pragma unroll
  for (int j = 0; j < 4; ++j) {
    int tr = lane * 4 + j;
    int r = tr >> 3, seg = tr & 7;
    async_b128(qstb + r * 144 + seg * 16,
               qgb + (unsigned long long)r * (CC * 2) + seg * 16);
  }
  wait_async0();
  __syncthreads();

  v16bf aq[2];
#pragma unroll
  for (int w = 0; w < 2; ++w)
#pragma unroll
    for (int i = 0; i < 16; ++i)
      aq[w][i] = bfbits(qst[l16][w * 32 + amapK(i, hi)]);

  const float scale = 0.125f;  // 64^-0.5

  // ---- scores ----
  for (int kt = 0; kt < 32; ++kt) {
    const int m0 = kt * 16;
    __syncthreads();
    const unsigned long long kgb =
        kvgb + (((unsigned long long)b * NN2 + m0) * C2 + h * DD) * 2;
#pragma unroll
    for (int j = 0; j < 4; ++j) {  // 16 keys x 128B
      int tr = lane * 4 + j;
      int r = tr >> 3, seg = tr & 7;
      async_b128(kvlb + r * 144 + seg * 16,
                 kgb + (unsigned long long)r * (C2 * 2) + seg * 16);
    }
    wait_async0();
    __syncthreads();
    v16bf bk[2];
#pragma unroll
    for (int w = 0; w < 2; ++w)
#pragma unroll
      for (int i = 0; i < 16; ++i)
        bk[w][i] = bfbits(kvst[l16][w * 32 + amapK(i, hi)]);
    v8f cf = {};
    cf = wmma_bf16(aq[0], bk[0], cf);
    cf = wmma_bf16(aq[1], bk[1], cf);
#pragma unroll
    for (int v = 0; v < 8; ++v)
      sc[v + hi * 8][m0 + l16] = cf[v] * scale;
  }
  __syncthreads();

  // ---- softmax (fp32) ----
  if (lane < 16) {
    float mx = -1e30f;
    for (int j = 0; j < 512; ++j) mx = fmaxf(mx, sc[lane][j]);
    float sum = 0.f;
    for (int j = 0; j < 512; ++j) {
      float e = __expf(sc[lane][j] - mx);
      sc[lane][j] = e;
      sum += e;
    }
    float inv = 1.0f / sum;
    for (int j = 0; j < 512; ++j) pl[lane][j] = f2bf(sc[lane][j] * inv);
  }
  __syncthreads();

  // ---- O = P @ V ----
  v8f o[4] = {};
  for (int kt = 0; kt < 16; ++kt) {
    const int m0 = kt * 32;
    __syncthreads();
    const unsigned long long vgb =
        kvgb + (((unsigned long long)b * NN2 + m0) * C2 + CC + h * DD) * 2;
#pragma unroll
    for (int j = 0; j < 8; ++j) {  // 32 keys x 128B
      int tr = lane * 8 + j;
      int r = tr >> 3, seg = tr & 7;
      async_b128(kvlb + r * 144 + seg * 16,
                 vgb + (unsigned long long)r * (C2 * 2) + seg * 16);
    }
    wait_async0();
    __syncthreads();
    v16bf ap;
#pragma unroll
    for (int i = 0; i < 16; ++i)
      ap[i] = bfbits(pl[l16][m0 + amapK(i, hi)]);
#pragma unroll
    for (int fn = 0; fn < 4; ++fn) {
      v16bf bv;
#pragma unroll
      for (int i = 0; i < 16; ++i)
        bv[i] = bfbits(kvst[amapK(i, hi)][fn * 16 + l16]);
      o[fn] = wmma_bf16(ap, bv, o[fn]);
    }
  }

#pragma unroll
  for (int fn = 0; fn < 4; ++fn)
#pragma unroll
    for (int v = 0; v < 8; ++v) {
      const int row = v + hi * 8;
      ao[((size_t)b * SS + s0 + row) * CC + h * DD + fn * 16 + l16] =
          f2bf(o[fn][v]);
    }
}

// ---------------------------------------------------------------------------
// adaLN MLP (64 rows, fp32 VALU — too small for WMMA to matter)
// ---------------------------------------------------------------------------
__global__ __launch_bounds__(256) void ada_kernel(
    const float* __restrict__ t, const float* __restrict__ c,
    const float* __restrict__ wada, const float* __restrict__ bada,
    float* __restrict__ ada) {
  __shared__ float al[CC];
  const int g = blockIdx.x;
  const int b = g >> 4;
  const int tid = threadIdx.x;
#pragma unroll
  for (int e = 0; e < 4; ++e) {
    int k = tid * 4 + e;
    float x = t[(size_t)b * CC + k] + c[(size_t)g * CC + k];
    al[k] = x / (1.0f + __expf(-x));
  }
  __syncthreads();
  float s[8];
#pragma unroll
  for (int e = 0; e < 8; ++e) s[e] = bada[tid * 8 + e];
  for (int k = 0; k < CC; ++k) {
    const float a = al[k];
    const float* wr = wada + (size_t)k * C2 + tid * 8;
#pragma unroll
    for (int e = 0; e < 8; ++e) s[e] = fmaf(a, wr[e], s[e]);
  }
#pragma unroll
  for (int e = 0; e < 8; ++e) ada[(size_t)g * C2 + tid * 8 + e] = s[e];
}

// ---------------------------------------------------------------------------
// LayerNorm + modulate + final linear (1024 -> 32) via WMMA.
// ---------------------------------------------------------------------------
__global__ __launch_bounds__(32) void lnfinal_kernel(
    const float* __restrict__ x2, const float* __restrict__ ada,
    const unsigned short* __restrict__ wlin, const float* __restrict__ blin,
    float* __restrict__ y) {
  __shared__ unsigned short ml[16][1032];
  __shared__ unsigned short wst[32][40];  // row 80 B

  const int lane = threadIdx.x;
  const int l16  = lane & 15;
  const int hi   = lane >> 4;
  const int rt = blockIdx.x;
  const int g  = blockIdx.y;
  const size_t rowbase = (size_t)g * NT + rt * 16;

  if (lane < 16) {
    const int r = lane;
    const float* xr = x2 + (rowbase + r) * CC;
    float s = 0.f, ss = 0.f;
    for (int k = 0; k < CC; k += 4) {
      float4 v4 = *(const float4*)(xr + k);
      s  += v4.x + v4.y + v4.z + v4.w;
      ss += v4.x * v4.x + v4.y * v4.y + v4.z * v4.z + v4.w * v4.w;
    }
    const float mu   = s * (1.0f / CC);
    const float var  = ss * (1.0f / CC) - mu * mu;
    const float rstd = rsqrtf(var + 1e-6f);
    const float* adag = ada + (size_t)g * C2;
    for (int k = 0; k < CC; ++k) {
      float ln = (xr[k] - mu) * rstd;
      float mo = ln * (1.0f + adag[CC + k]) + adag[k];
      ml[r][k] = f2bf(mo);
    }
  }
  __syncthreads();

  const unsigned wstb = lds_off(&wst[0][0]);
  const unsigned long long wgb = (unsigned long long)(uintptr_t)wlin;

  v8f o[2] = {};
  for (int kc = 0; kc < 32; ++kc) {
    __syncthreads();
#pragma unroll
    for (int j = 0; j < 4; ++j) {  // 32 rows x 64B : 128 transfers, 4/lane
      int tr = lane * 4 + j;
      int r = tr >> 2, seg = tr & 3;
      async_b128(wstb + r * 80 + seg * 16,
                 wgb + (unsigned long long)(kc * 32 + r) * (OUTD * 2) + seg * 16);
    }
    wait_async0();
    __syncthreads();
    v16bf a;
#pragma unroll
    for (int i = 0; i < 16; ++i)
      a[i] = bfbits(ml[l16][kc * 32 + amapK(i, hi)]);
#pragma unroll
    for (int fn = 0; fn < 2; ++fn) {
      v16bf bb;
#pragma unroll
      for (int i = 0; i < 16; ++i)
        bb[i] = bfbits(wst[amapK(i, hi)][fn * 16 + l16]);
      o[fn] = wmma_bf16(a, bb, o[fn]);
    }
  }
#pragma unroll
  for (int fn = 0; fn < 2; ++fn)
#pragma unroll
    for (int v = 0; v < 8; ++v) {
      const int row = v + hi * 8;
      y[(rowbase + row) * OUTD + fn * 16 + l16] =
          o[fn][v] + blin[fn * 16 + l16];
    }
}

// ---------------------------------------------------------------------------
// Host launcher
// ---------------------------------------------------------------------------
extern "C" void kernel_launch(void* const* d_in, const int* in_sizes, int n_in,
                              void* d_out, int out_size, void* d_ws,
                              size_t ws_size, hipStream_t stream) {
  (void)in_sizes; (void)n_in; (void)out_size; (void)ws_size;
  const float* x     = (const float*)d_in[0];
  const float* v     = (const float*)d_in[1];
  const float* t     = (const float*)d_in[2];
  const float* c     = (const float*)d_in[3];
  const float* wq    = (const float*)d_in[4];
  const float* bq    = (const float*)d_in[5];
  const float* wkv   = (const float*)d_in[6];
  const float* bkv   = (const float*)d_in[7];
  const float* wproj = (const float*)d_in[8];
  const float* bproj = (const float*)d_in[9];
  const float* wada  = (const float*)d_in[10];
  const float* bada  = (const float*)d_in[11];
  const float* wlin  = (const float*)d_in[12];
  const float* blin  = (const float*)d_in[13];
  float* y = (float*)d_out;

  const size_t M1 = (size_t)BB * SS;  // 16384
  char* ws = (char*)d_ws;
  unsigned short* xb     = (unsigned short*)ws;               ws += M1 * CC * 2;
  unsigned short* vb     = (unsigned short*)ws;               ws += (size_t)BB * NN2 * CC * 2;
  unsigned short* wqb    = (unsigned short*)ws;               ws += (size_t)CC * CC * 2;
  unsigned short* wkvb   = (unsigned short*)ws;               ws += (size_t)CC * C2 * 2;
  unsigned short* wprojb = (unsigned short*)ws;               ws += (size_t)CC * CC * 2;
  unsigned short* wlinb  = (unsigned short*)ws;               ws += (size_t)CC * OUTD * 2;
  unsigned short* q_ws   = (unsigned short*)ws;               ws += M1 * CC * 2;
  unsigned short* kv_ws  = (unsigned short*)ws;               ws += (size_t)BB * NN2 * C2 * 2;
  unsigned short* ao_ws  = (unsigned short*)ws;               ws += M1 * CC * 2;
  float*          x2_ws  = (float*)ws;                        ws += M1 * CC * 4;
  float*          ada_ws = (float*)ws;

  auto cvt = [&](const float* src, unsigned short* dst, size_t n) {
    int n8 = (int)(n / 8);
    cvt_kernel<<<(n8 + 255) / 256, 256, 0, stream>>>(src, dst, n8);
  };
  cvt(x, xb, M1 * CC);
  cvt(v, vb, (size_t)BB * NN2 * CC);
  cvt(wq, wqb, (size_t)CC * CC);
  cvt(wkv, wkvb, (size_t)CC * C2);
  cvt(wproj, wprojb, (size_t)CC * CC);
  cvt(wlin, wlinb, (size_t)CC * OUTD);

  // 1) q = x @ wq + bq  (-> bf16)
  gemm_wmma_kernel<false, true>
      <<<dim3(CC / 64, (int)(M1 / 128)), 256, 0, stream>>>(
          xb, wqb, bq, nullptr, q_ws, (int)M1, CC, CC);

  // 2) kv = v @ wkv + bkv (-> bf16)
  gemm_wmma_kernel<false, true>
      <<<dim3(C2 / 64, (BB * NN2) / 128), 256, 0, stream>>>(
          vb, wkvb, bkv, nullptr, kv_ws, BB * NN2, C2, CC);

  // 3) attention -> ao (bf16)
  attn_kernel<<<dim3(SS / 16, HH, BB), 32, 0, stream>>>(q_ws, kv_ws, ao_ws);

  // 4) x2 = x + ao @ wproj + bproj (fp32)
  gemm_wmma_kernel<true, false>
      <<<dim3(CC / 64, (int)(M1 / 128)), 256, 0, stream>>>(
          ao_ws, wprojb, bproj, x, x2_ws, (int)M1, CC, CC);

  // 5) ada = silu(t + c) @ wada + bada
  ada_kernel<<<dim3(BB * TT), 256, 0, stream>>>(t, c, wada, bada, ada_ws);

  // 6) LN + modulate + final linear -> y
  lnfinal_kernel<<<dim3(NT / 16, BB * TT), 32, 0, stream>>>(
      x2_ws, ada_ws, wlinb, blin, y);
}
